// CausalSelfAttention_34832184770863
// MI455X (gfx1250) — matrix operand order repaired and checked by reading, not compiled, not run on hardware
//
#include <hip/hip_runtime.h>

// ---------------------------------------------------------------------------
// CausalSelfAttention (quirky: scale=sqrt(D_OUT)=32, softmax over QUERY axis)
// B=2, S=2048, D=1024, H=16, HD=64.  All matmuls on v_wmma_f32_16x16x32_f16.
// QKV GEMM stages tiles with GLOBAL_LOAD_ASYNC_TO_LDS_B128 (ASYNCcnt,
// double-buffered) per CDNA5 ISA.
// ---------------------------------------------------------------------------

typedef __attribute__((ext_vector_type(16))) _Float16 v16h;
typedef __attribute__((ext_vector_type(8)))  float    v8f;
typedef __attribute__((ext_vector_type(4)))  unsigned int u32x4;

#define S_LEN 2048
#define B_SZ  2
#define H_CNT 16
#define HD_   64
#define DM    1024
#define NTOK  (B_SZ * S_LEN)          // 4096
#define QTILES (S_LEN / 16)           // 128
#define INV_SCALE 0.03125f            // 1/sqrt(1024)

// ---- WMMA operand loaders (layouts per CDNA5 ISA 7.12.2, wave32) ----------
// A (16x32 f16): lane L -> row m=L&15, half h=L>>4 holds K = {8h..8h+7} in
// v[0..7] and K = {16+8h..16+8h+7} in v[8..15]  -> two 16B loads.
__device__ __forceinline__ v16h loadA(const _Float16* p, int stride, int m, int h) {
  union { v16h v; u32x4 q[2]; } u;
  const _Float16* r = p + (size_t)m * stride;
  u.q[0] = *(const u32x4*)(r + 8 * h);
  u.q[1] = *(const u32x4*)(r + 16 + 8 * h);
  return u.v;
}
// B (32x16 f16) from an N-major buffer (row = N, K contiguous): lane L ->
// column n=L&15, half h=L>>4 holds K = {8h..8h+7} in v[0..7] and
// K = {16+8h..16+8h+7} in v[8..15] -> two 16B loads.
__device__ __forceinline__ v16h loadB(const _Float16* p, int stride, int n, int h) {
  union { v16h v; u32x4 q[2]; } u;
  const _Float16* r = p + (size_t)n * stride;
  u.q[0] = *(const u32x4*)(r + 8 * h);
  u.q[1] = *(const u32x4*)(r + 16 + 8 * h);
  return u.v;
}
__device__ __forceinline__ v8f wmma16(v16h a, v16h b, v8f c) {
  return __builtin_amdgcn_wmma_f32_16x16x32_f16(false, a, false, b, (short)0, c,
                                                false, false);
}
// Generic pointers to LDS objects are {shared_aperture, as3_offset}; the low
// 32 bits are exactly the LDS address the async-LDS instructions expect.
__device__ __forceinline__ unsigned lds_off(const void* p) {
  return (unsigned)(uintptr_t)p;
}

// ---- phase 0: fp32 -> f16 conversions -------------------------------------
__global__ void cvt_f32_to_f16(const float* __restrict__ in,
                               _Float16* __restrict__ out, int n) {
  int i = blockIdx.x * blockDim.x + threadIdx.x;
  if (i < n) out[i] = (_Float16)in[i];
}
// W is [K=d_in][N=d_out]; emit Wt[N][K] (f16) so GEMM B operand is K-contiguous.
__global__ void cvt_transpose_W(const float* __restrict__ W,
                                _Float16* __restrict__ Wt) {
  int i = blockIdx.x * blockDim.x + threadIdx.x;   // over DM*DM
  int nrow = i >> 10, kcol = i & (DM - 1);
  Wt[i] = (_Float16)W[(size_t)kcol * DM + nrow];
}

// ---- phase 1: y = x @ W  (64x128 block tile, 8 waves) ---------------------
// Tile staging via async global->LDS copies, double buffered on ASYNCcnt.
__global__ void __launch_bounds__(256)
gemm_qkv_kernel(const _Float16* __restrict__ xa, const _Float16* __restrict__ Wt,
                _Float16* __restrict__ outp, int transpose_out) {
  __shared__ alignas(16) _Float16 As[2][64 * 32];
  __shared__ alignas(16) _Float16 Bs[2][128 * 32];
  int tid = threadIdx.x;
  int lane = tid & 31, w = tid >> 5;
  int n = lane & 15, h2 = lane >> 4;
  int wm = w & 3, wn = w >> 2;                 // 4 (M) x 2 (N) waves
  int bm = blockIdx.x * 64, bn = blockIdx.y * 128;

  // per-thread 16B staging chunks: A = 64x32 (one chunk), B = 128x32 (two)
  int ia = tid * 8;           int ra  = ia  >> 5, ca  = ia  & 31;
  int ib1 = (tid + 256) * 8;  int rb1 = ib1 >> 5, cb1 = ib1 & 31;
  unsigned long long xa64 = (unsigned long long)xa;
  unsigned long long wt64 = (unsigned long long)Wt;

  auto issue = [&](int kk, int buf) {
    unsigned la  = lds_off(&As[buf][ia]);
    unsigned lb0 = lds_off(&Bs[buf][ia]);
    unsigned lb1 = lds_off(&Bs[buf][ib1]);
    unsigned va  = (unsigned)(((bm + ra)  * DM + kk + ca)  * 2);
    unsigned vb0 = (unsigned)(((bn + ra)  * DM + kk + ca)  * 2);
    unsigned vb1 = (unsigned)(((bn + rb1) * DM + kk + cb1) * 2);
    asm volatile(
        "global_load_async_to_lds_b128 %0, %1, %6\n\t"
        "global_load_async_to_lds_b128 %2, %3, %7\n\t"
        "global_load_async_to_lds_b128 %4, %5, %7"
        :: "v"(la), "v"(va), "v"(lb0), "v"(vb0), "v"(lb1), "v"(vb1),
           "s"(xa64), "s"(wt64)
        : "memory");
  };

  issue(0, 0);
  issue(32, 1);

  v8f a0 = {}, a1 = {}, a2 = {}, a3 = {};
  for (int kk = 0; kk < DM; kk += 32) {
    int cur = (kk >> 5) & 1;
    // Steady state: 3 newer outstanding async ops belong to the other buffer
    // (async loads complete in order); final tile has nothing newer.
    if (kk + 32 < DM) asm volatile("s_wait_asynccnt 3" ::: "memory");
    else              asm volatile("s_wait_asynccnt 0" ::: "memory");
    __syncthreads();                               // all waves' data visible
    v16h Av = loadA(&As[cur][wm * 16 * 32], 32, n, h2);
    a0 = wmma16(Av, loadB(&Bs[cur][(wn * 64 +  0) * 32], 32, n, h2), a0);
    a1 = wmma16(Av, loadB(&Bs[cur][(wn * 64 + 16) * 32], 32, n, h2), a1);
    a2 = wmma16(Av, loadB(&Bs[cur][(wn * 64 + 32) * 32], 32, n, h2), a2);
    a3 = wmma16(Av, loadB(&Bs[cur][(wn * 64 + 48) * 32], 32, n, h2), a3);
    __syncthreads();                               // done reading buf[cur]
    if (kk + 64 < DM) issue(kk + 64, cur);
  }

#pragma unroll
  for (int dt = 0; dt < 4; ++dt) {
    v8f a = (dt == 0 ? a0 : dt == 1 ? a1 : dt == 2 ? a2 : a3);
#pragma unroll
    for (int i = 0; i < 8; ++i) {     // C layout: row = 8*h2+i, col = n
      int row = bm + wm * 16 + 8 * h2 + i;
      int col = bn + wn * 64 + dt * 16 + n;
      if (!transpose_out) {
        outp[(size_t)row * DM + col] = (_Float16)a[i];
      } else {                         // v stored as vT[b][h][hd][s]
        int b = row >> 11, s = row & (S_LEN - 1);
        int hh = col >> 6, d = col & 63;
        outp[(((size_t)(b * H_CNT + hh)) * HD_ + d) * S_LEN + s] = (_Float16)a[i];
      }
    }
  }
}

// ---- phase 2a: per key-column max & sum-exp over q >= k (online) ----------
__global__ void colstats_kernel(const _Float16* __restrict__ qb,
                                const _Float16* __restrict__ kf,
                                float* __restrict__ colmax,
                                float* __restrict__ colsum) {
  int bid = blockIdx.x;
  int kt = bid & (QTILES - 1);
  int head = (bid >> 7) & (H_CNT - 1);
  int b = bid >> 11;
  int lane = threadIdx.x;
  int n = lane & 15, h2 = lane >> 4;
  int c0 = kt * 16;

  const _Float16* kbase = kf + ((size_t)(b * S_LEN + c0)) * DM + head * HD_;
  v16h Bk0 = loadB(kbase, DM, n, h2);        // hd 0..31
  v16h Bk1 = loadB(kbase + 32, DM, n, h2);   // hd 32..63

  float M = -__builtin_inff();
  float Ssum = 0.f;

  for (int qt = kt; qt < QTILES; ++qt) {     // only q >= k contributes
    int q0 = qt * 16;
    const _Float16* qbase = qb + ((size_t)(b * S_LEN + q0)) * DM + head * HD_;
    v16h A0 = loadA(qbase, DM, n, h2);
    v16h A1 = loadA(qbase + 32, DM, n, h2);
    v8f z = {};
    v8f sc = wmma16(A1, Bk1, wmma16(A0, Bk0, z));

    float tv[8], tm = -__builtin_inff();
#pragma unroll
    for (int i = 0; i < 8; ++i) {
      int q = q0 + 8 * h2 + i, k = c0 + n;
      float t = (q >= k) ? sc[i] * INV_SCALE : -__builtin_inff();
      tv[i] = t;
      tm = fmaxf(tm, t);
    }
    tm = fmaxf(tm, __shfl_xor(tm, 16, 32));  // combine the two row-halves
    float Mn = fmaxf(M, tm);                 // Mn finite after diagonal tile
    float corr = __expf(M - Mn);
    float ps = 0.f;
#pragma unroll
    for (int i = 0; i < 8; ++i) ps += __expf(tv[i] - Mn);
    ps += __shfl_xor(ps, 16, 32);
    Ssum = Ssum * corr + ps;
    M = Mn;
  }
  if (lane < 16) {
    size_t cb = ((size_t)(b * H_CNT + head)) * S_LEN + c0 + n;
    colmax[cb] = M;
    colsum[cb] = Ssum;
  }
}

// ---- phase 2b: ctx[q,:] = sum_k exp(t-m[k])/denom[k] * v[k,:] -------------
__global__ void attn_out_kernel(const _Float16* __restrict__ qb,
                                const _Float16* __restrict__ kf,
                                const _Float16* __restrict__ vT,
                                const float* __restrict__ colmax,
                                const float* __restrict__ colsum,
                                float* __restrict__ outp) {
  __shared__ alignas(16) _Float16 wtile[16 * 32];  // p tile, row-major [q][k]
  int bid = blockIdx.x;
  int qt = bid & (QTILES - 1);
  int head = (bid >> 7) & (H_CNT - 1);
  int b = bid >> 11;
  int lane = threadIdx.x;
  int n = lane & 15, h2 = lane >> 4;
  int q0 = qt * 16;

  const _Float16* qbase = qb + ((size_t)(b * S_LEN + q0)) * DM + head * HD_;
  v16h A0 = loadA(qbase, DM, n, h2);
  v16h A1 = loadA(qbase + 32, DM, n, h2);

  v8f c0a = {}, c1a = {}, c2a = {}, c3a = {};
  size_t cbase = ((size_t)(b * H_CNT + head)) * S_LEN;
  const _Float16* vbase = vT + ((size_t)(b * H_CNT + head)) * HD_ * S_LEN;

  for (int kblk = 0; kblk <= q0 + 15; kblk += 32) {
#pragma unroll
    for (int half = 0; half < 2; ++half) {
      int kc0 = kblk + half * 16;
      const _Float16* kbase = kf + ((size_t)(b * S_LEN + kc0)) * DM + head * HD_;
      v16h Bk0 = loadB(kbase, DM, n, h2);
      v16h Bk1 = loadB(kbase + 32, DM, n, h2);
      v8f z = {};
      v8f sc = wmma16(A1, Bk1, wmma16(A0, Bk0, z));
      float cm = colmax[cbase + kc0 + n];
      float ri = 1.0f / colsum[cbase + kc0 + n];
#pragma unroll
      for (int i = 0; i < 8; ++i) {
        int q = q0 + 8 * h2 + i, k = kc0 + n;
        float t = (q >= k) ? sc[i] * INV_SCALE : -__builtin_inff();
        float p = __expf(t - cm) * ri;
        wtile[(8 * h2 + i) * 32 + half * 16 + n] = (_Float16)p;  // [q][k]
      }
    }
    asm volatile("s_wait_dscnt 0" ::: "memory");   // cross-lane LDS exchange
    v16h Aw = loadA(wtile, 32, n, h2);             // p as WMMA-A (16x32)
    c0a = wmma16(Aw, loadB(vbase + (size_t)( 0) * S_LEN + kblk, S_LEN, n, h2), c0a);
    c1a = wmma16(Aw, loadB(vbase + (size_t)(16) * S_LEN + kblk, S_LEN, n, h2), c1a);
    c2a = wmma16(Aw, loadB(vbase + (size_t)(32) * S_LEN + kblk, S_LEN, n, h2), c2a);
    c3a = wmma16(Aw, loadB(vbase + (size_t)(48) * S_LEN + kblk, S_LEN, n, h2), c3a);
  }
#pragma unroll
  for (int dt = 0; dt < 4; ++dt) {
    v8f a = (dt == 0 ? c0a : dt == 1 ? c1a : dt == 2 ? c2a : c3a);
#pragma unroll
    for (int i = 0; i < 8; ++i) {
      int row = q0 + 8 * h2 + i;
      int col = head * 64 + dt * 16 + n;
      outp[((size_t)(b * S_LEN + row)) * DM + col] = a[i];
    }
  }
}

// ---------------------------------------------------------------------------
extern "C" void kernel_launch(void* const* d_in, const int* in_sizes, int n_in,
                              void* d_out, int out_size, void* d_ws, size_t ws_size,
                              hipStream_t stream) {
  const float* x  = (const float*)d_in[0];
  const float* Wq = (const float*)d_in[1];
  const float* Wk = (const float*)d_in[2];
  const float* Wv = (const float*)d_in[3];
  float* outp = (float*)d_out;

  char* ws = (char*)d_ws;                         // ~38.5 MB used
  _Float16* xb   = (_Float16*)(ws);               // 8 MB  [4096][1024]
  _Float16* Wtq  = (_Float16*)(ws + ( 8u << 20)); // 2 MB  [N][K]
  _Float16* Wtk  = (_Float16*)(ws + (10u << 20));
  _Float16* Wtv  = (_Float16*)(ws + (12u << 20));
  _Float16* qb   = (_Float16*)(ws + (14u << 20)); // 8 MB  [b][s][h][hd]
  _Float16* kb_  = (_Float16*)(ws + (22u << 20)); // 8 MB  [b][s][h][hd]
  _Float16* vTb  = (_Float16*)(ws + (30u << 20)); // 8 MB  [b][h][hd][s]
  float* colmax  = (float*)(ws + (38u << 20));            // 256 KB
  float* colsum  = (float*)(ws + (38u << 20) + (1u << 18)); // 256 KB

  cvt_f32_to_f16<<<(NTOK * DM) / 256, 256, 0, stream>>>(x, xb, NTOK * DM);
  cvt_transpose_W<<<(DM * DM) / 256, 256, 0, stream>>>(Wq, Wtq);
  cvt_transpose_W<<<(DM * DM) / 256, 256, 0, stream>>>(Wk, Wtk);
  cvt_transpose_W<<<(DM * DM) / 256, 256, 0, stream>>>(Wv, Wtv);

  dim3 gg(NTOK / 64, DM / 128);
  gemm_qkv_kernel<<<gg, 256, 0, stream>>>(xb, Wtq, qb, 0);
  gemm_qkv_kernel<<<gg, 256, 0, stream>>>(xb, Wtk, kb_, 0);
  gemm_qkv_kernel<<<gg, 256, 0, stream>>>(xb, Wtv, vTb, 1);

  int nblk = B_SZ * H_CNT * QTILES;               // 4096 single-wave blocks
  colstats_kernel<<<nblk, 32, 0, stream>>>(qb, kb_, colmax, colsum);
  attn_out_kernel<<<nblk, 32, 0, stream>>>(qb, kb_, vTb, colmax, colsum, outp);
}
